// GNNProcessor_GAT_60687887892632
// MI455X (gfx1250) — compile-verified, matrix operand order checked by reading, not run on hardware
//
#include <hip/hip_runtime.h>
#include <hip/hip_bf16.h>

#define NNODES 50000
#define NEDGES 800000
#define HDIM   64
#define NHEADS 4
#define EPSLN  1e-5f
#define NSLOPE 0.2f

typedef __attribute__((ext_vector_type(16))) __bf16       bf16x16;
typedef __attribute__((ext_vector_type(8)))  float        v8f;
typedef __attribute__((ext_vector_type(8)))  unsigned int u32x8;

static __device__ __forceinline__ float leaky(float v) {
    return v > 0.0f ? v : NSLOPE * v;
}

// ---------------------------------------------------------------- utilities

__global__ void fill_f32(float* p, float v, long n) {
    long i = (long)blockIdx.x * blockDim.x + threadIdx.x;
    if (i < n) p[i] = v;
}

__global__ void f32_to_bf16(const float* __restrict__ x, __bf16* __restrict__ xb, long n) {
    long i = (long)blockIdx.x * blockDim.x + threadIdx.x;
    if (i < n) xb[i] = (__bf16)x[i];
}

// Wt[n][k] = W[k][n]  (so both GEMM operands are K-contiguous)
__global__ void transpose_w_bf16(const float* __restrict__ W, __bf16* __restrict__ Wt,
                                 int K, int Nout) {
    int i = blockIdx.x * blockDim.x + threadIdx.x;
    if (i >= K * Nout) return;
    int k = i / Nout, n = i % Nout;
    Wt[(long)n * K + k] = (__bf16)W[i];
}

// ------------------------------------------------- initial scatter-mean of edges

__global__ void count_edges(const int* __restrict__ col, float* cnt, int E) {
    int i = blockIdx.x * blockDim.x + threadIdx.x;
    if (i < E) atomicAdd(&cnt[col[i]], 1.0f);
}

__global__ void scatter_edge_embeds(const int* __restrict__ col, const float* __restrict__ ee,
                                    float* acc, int E) {
    long i = (long)blockIdx.x * blockDim.x + threadIdx.x;
    if (i >= (long)E * HDIM) return;
    int e = (int)(i >> 6);
    atomicAdd(&acc[(long)col[e] * HDIM + (i & 63)], ee[i]);
}

__global__ void init_x(const float* __restrict__ node, const float* __restrict__ acc,
                       const float* __restrict__ cnt, float* __restrict__ x) {
    long i = (long)blockIdx.x * blockDim.x + threadIdx.x;
    if (i >= (long)NNODES * HDIM) return;
    float c = cnt[i >> 6];
    c = c < 1.0f ? 1.0f : c;
    x[i] = node[i] + acc[i] / c;
}

// ------------------------------------------------------------- WMMA GEMM
// C[M x Nout] = A[M x K] * B[K x Nout], A bf16 row-major, Bt = B^T bf16 (Nout x K).
// One wave computes a 16x64 tile: 4 accumulators share one A fragment, so A
// (L2-resident activations) is fetched once per 4 wmmas.  Nout % 64 == 0.
// 16-bit A/B layout (ISA 7.12.2): lane<16 -> K base 0, lane>=16 -> K base 8;
// VGPR pair p holds K = kb + (p>=4 ? 16 : 0) + 2*(p&3) .. +1 (contiguous dword).
__global__ void gemm_wmma_bf16(const __bf16* __restrict__ A, const __bf16* __restrict__ Bt,
                               float* __restrict__ C, int M, int K, int Nout) {
    int gwave = (int)(((long)blockIdx.x * blockDim.x + threadIdx.x) >> 5);
    int lane  = threadIdx.x & 31;
    int groupsN = Nout >> 6;              // 16x64 tiles along N
    int tilesM  = M >> 4;
    if (gwave >= tilesM * groupsN) return;   // wave-uniform: EXEC stays all-ones
    int tm = gwave / groupsN;
    int gn = gwave % groupsN;

    int arow = tm * 16 + (lane & 15);
    int kb   = (lane < 16) ? 0 : 8;
    const __bf16* ap = A  + (long)arow * K;
    const __bf16* bp = Bt + (long)(gn * 64 + (lane & 15)) * K;

    v8f acc0 = {}, acc1 = {}, acc2 = {}, acc3 = {};
    for (int k0 = 0; k0 < K; k0 += 32) {
        u32x8 av, bv0, bv1, bv2, bv3;
#pragma unroll
        for (int p = 0; p < 8; ++p) {
            int off = k0 + kb + ((p & 4) ? 16 : 0) + 2 * (p & 3);
            av[p]  = *(const unsigned int*)(ap + off);
            bv0[p] = *(const unsigned int*)(bp + off);
            bv1[p] = *(const unsigned int*)(bp + 16 * K + off);
            bv2[p] = *(const unsigned int*)(bp + 32 * K + off);
            bv3[p] = *(const unsigned int*)(bp + 48 * K + off);
        }
        bf16x16 a = __builtin_bit_cast(bf16x16, av);
        acc0 = __builtin_amdgcn_wmma_f32_16x16x32_bf16(false, a, false,
                   __builtin_bit_cast(bf16x16, bv0), (short)0, acc0, false, false);
        acc1 = __builtin_amdgcn_wmma_f32_16x16x32_bf16(false, a, false,
                   __builtin_bit_cast(bf16x16, bv1), (short)0, acc1, false, false);
        acc2 = __builtin_amdgcn_wmma_f32_16x16x32_bf16(false, a, false,
                   __builtin_bit_cast(bf16x16, bv2), (short)0, acc2, false, false);
        acc3 = __builtin_amdgcn_wmma_f32_16x16x32_bf16(false, a, false,
                   __builtin_bit_cast(bf16x16, bv3), (short)0, acc3, false, false);
    }

    // C/D layout: VGPR v, lane l -> M = v + (l>=16 ? 8 : 0), N = l&15
    int cm = tm * 16 + ((lane & 16) ? 8 : 0);
    int cn = gn * 64 + (lane & 15);
    float* cp = C + (long)cm * Nout + cn;
#pragma unroll
    for (int v = 0; v < 8; ++v) {
        long o = (long)v * Nout;
        cp[o +  0] = acc0[v];
        cp[o + 16] = acc1[v];
        cp[o + 32] = acc2[v];
        cp[o + 48] = acc3[v];
    }
}

// ----------------------------------------------------- attention coefficients

__global__ void attn_coef(const float* __restrict__ h, const float* __restrict__ asrc,
                          const float* __restrict__ adst, float* __restrict__ as_,
                          float* __restrict__ ad_, int heads) {
    int i = blockIdx.x * blockDim.x + threadIdx.x;   // i = n*heads + head
    if (i >= NNODES * heads) return;
    int hd = i % heads;
    const float* hp = h + (long)i * HDIM;
    const float* sp = asrc + hd * HDIM;
    const float* dp = adst + hd * HDIM;
    float s = 0.0f, d = 0.0f;
#pragma unroll
    for (int k = 0; k < HDIM; ++k) { float v = hp[k]; s += v * sp[k]; d += v * dp[k]; }
    as_[i] = s;
    ad_[i] = d;
}

// --------------------------------------------------------------- edge passes

static __device__ __forceinline__ void edge_rc(int e, const int* row, const int* col,
                                               int* r, int* c) {
    if (e < NEDGES) { *r = row[e]; *c = col[e]; }
    else            { *r = e - NEDGES; *c = e - NEDGES; }   // self loops
}

static __device__ __forceinline__ void atomicMaxF(float* addr, float v) {
    unsigned int bits = __float_as_uint(v);
    if ((int)bits >= 0) atomicMax((int*)addr, (int)bits);
    else                atomicMin((unsigned int*)addr, bits);
}

__global__ void edge_max_k(const int* __restrict__ row, const int* __restrict__ col,
                           const float* __restrict__ as_, const float* __restrict__ ad_,
                           float* m, int heads) {
    long i = (long)blockIdx.x * blockDim.x + threadIdx.x;
    if (i >= (long)(NEDGES + NNODES) * heads) return;
    int e = (int)(i / heads), hd = (int)(i % heads);
    int r, c; edge_rc(e, row, col, &r, &c);
    float v = leaky(as_[r * heads + hd] + ad_[c * heads + hd]);
    atomicMaxF(&m[c * heads + hd], v);
}

__global__ void edge_expsum_k(const int* __restrict__ row, const int* __restrict__ col,
                              const float* __restrict__ as_, const float* __restrict__ ad_,
                              const float* __restrict__ m, float* s, int heads) {
    long i = (long)blockIdx.x * blockDim.x + threadIdx.x;
    if (i >= (long)(NEDGES + NNODES) * heads) return;
    int e = (int)(i / heads), hd = (int)(i % heads);
    int r, c; edge_rc(e, row, col, &r, &c);
    int ch = c * heads + hd;
    float v = leaky(as_[r * heads + hd] + ad_[ch]);
    atomicAdd(&s[ch], expf(v - m[ch]));
}

// alpha[e*heads+hd] = softmax weight (materialized once; 64 consumers each)
__global__ void edge_alpha_k(const int* __restrict__ row, const int* __restrict__ col,
                             const float* __restrict__ as_, const float* __restrict__ ad_,
                             const float* __restrict__ m, const float* __restrict__ s,
                             float* __restrict__ alpha, int heads) {
    long i = (long)blockIdx.x * blockDim.x + threadIdx.x;
    if (i >= (long)(NEDGES + NNODES) * heads) return;
    int e = (int)(i / heads), hd = (int)(i % heads);
    int r, c; edge_rc(e, row, col, &r, &c);
    int ch = c * heads + hd;
    alpha[i] = expf(leaky(as_[r * heads + hd] + ad_[ch]) - m[ch]) / s[ch];
}

// thread per (edge, head, dim): coalesced h reads + coalesced float atomics
__global__ void edge_agg_k(const int* __restrict__ row, const int* __restrict__ col,
                           const float* __restrict__ alpha, const float* __restrict__ h,
                           float* agg, int heads) {
    long i = (long)blockIdx.x * blockDim.x + threadIdx.x;
    if (i >= (long)(NEDGES + NNODES) * heads * HDIM) return;
    int  d  = (int)(i & 63);
    long eh = i >> 6;                       // e*heads + hd
    int  hd = (int)(eh % heads);
    int  e  = (int)(eh / heads);
    int r, c; edge_rc(e, row, col, &r, &c);
    float a = alpha[eh];
    float v = h[((long)r * heads + hd) * HDIM + d] * a;
    atomicAdd(&agg[((long)c * heads + hd) * HDIM + d], v);
}

// ------------------------------------------------- bias + LayerNorm + ELU

__global__ void finalize_ln_elu(const float* __restrict__ agg, const float* __restrict__ bias,
                                const float* __restrict__ g, const float* __restrict__ be,
                                float* __restrict__ dst, int odim) {
    __shared__ float red[256];
    int n = blockIdx.x, t = threadIdx.x;
    float v = 0.0f;
    if (t < odim) v = agg[(long)n * odim + t] + bias[t];
    red[t] = v;
    __syncthreads();
    for (int s = 128; s > 0; s >>= 1) { if (t < s) red[t] += red[t + s]; __syncthreads(); }
    float mu = red[0] / (float)odim;
    __syncthreads();
    float dv = (t < odim) ? (v - mu) : 0.0f;
    red[t] = dv * dv;
    __syncthreads();
    for (int s = 128; s > 0; s >>= 1) { if (t < s) red[t] += red[t + s]; __syncthreads(); }
    float var = red[0] / (float)odim;
    if (t < odim) {
        float y = dv * rsqrtf(var + EPSLN) * g[t] + be[t];
        y = y > 0.0f ? y : expf(y) - 1.0f;                  // ELU
        dst[(long)n * odim + t] = y;
    }
}

// ------------------------------------------------------------------- launch

static inline unsigned gb(long n) { return (unsigned)((n + 255) / 256); }

extern "C" void kernel_launch(void* const* d_in, const int* in_sizes, int n_in,
                              void* d_out, int out_size, void* d_ws, size_t ws_size,
                              hipStream_t stream) {
    (void)in_sizes; (void)n_in; (void)out_size; (void)ws_size;

    const float* node = (const float*)d_in[0];
    const int*   ei   = (const int*)d_in[1];
    const float* eemb = (const float*)d_in[2];
    const int* row = ei;
    const int* col = ei + NEDGES;

    struct Layer { const float *W, *as, *ad, *b, *g, *be; int din, heads, odim; };
    Layer L[3] = {
        { (const float*)d_in[3],  (const float*)d_in[4],  (const float*)d_in[5],
          (const float*)d_in[6],  (const float*)d_in[7],  (const float*)d_in[8],
          HDIM, NHEADS, NHEADS * HDIM },
        { (const float*)d_in[9],  (const float*)d_in[10], (const float*)d_in[11],
          (const float*)d_in[12], (const float*)d_in[13], (const float*)d_in[14],
          NHEADS * HDIM, NHEADS, NHEADS * HDIM },
        { (const float*)d_in[15], (const float*)d_in[16], (const float*)d_in[17],
          (const float*)d_in[18], (const float*)d_in[19], (const float*)d_in[20],
          NHEADS * HDIM, 1, HDIM },
    };

    // ---- carve workspace
    char* ws = (char*)d_ws;
    size_t off = 0;
    auto carve = [&](size_t bytes) -> char* {
        char* p = ws + off;
        off += (bytes + 255) & ~(size_t)255;
        return p;
    };
    float*  xf   = (float*)carve((size_t)NNODES * 256 * 4);
    float*  hbuf = (float*)carve((size_t)NNODES * 256 * 4);
    float*  agg  = (float*)carve((size_t)NNODES * 256 * 4);
    __bf16* xb   = (__bf16*)carve((size_t)NNODES * 256 * 2);
    __bf16* Wt   = (__bf16*)carve((size_t)256 * 256 * 2);
    float*  asb  = (float*)carve((size_t)NNODES * NHEADS * 4);
    float*  adb  = (float*)carve((size_t)NNODES * NHEADS * 4);
    float*  mb   = (float*)carve((size_t)NNODES * NHEADS * 4);
    float*  sb   = (float*)carve((size_t)NNODES * NHEADS * 4);
    float*  alphab = (float*)carve((size_t)(NEDGES + NNODES) * NHEADS * 4);
    float*  cnt  = (float*)carve((size_t)NNODES * 4);

    // ---- x = node_embeds + scatter_mean(edge_embeds, col)
    long nx0 = (long)NNODES * HDIM;
    fill_f32<<<gb(nx0), 256, 0, stream>>>(agg, 0.0f, nx0);
    fill_f32<<<gb(NNODES), 256, 0, stream>>>(cnt, 0.0f, NNODES);
    count_edges<<<gb(NEDGES), 256, 0, stream>>>(col, cnt, NEDGES);
    scatter_edge_embeds<<<gb((long)NEDGES * HDIM), 256, 0, stream>>>(col, eemb, agg, NEDGES);
    init_x<<<gb(nx0), 256, 0, stream>>>(node, agg, cnt, xf);

    // ---- GAT layers
    for (int li = 0; li < 3; ++li) {
        const Layer& l = L[li];
        long nx  = (long)NNODES * l.din;
        long no  = (long)NNODES * l.odim;
        long nh  = (long)NNODES * l.heads;
        long te  = (long)(NEDGES + NNODES) * l.heads;

        f32_to_bf16<<<gb(nx), 256, 0, stream>>>(xf, xb, nx);
        transpose_w_bf16<<<gb((long)l.din * l.odim), 256, 0, stream>>>(l.W, Wt, l.din, l.odim);

        fill_f32<<<gb(no), 256, 0, stream>>>(agg, 0.0f, no);
        fill_f32<<<gb(nh), 256, 0, stream>>>(sb, 0.0f, nh);
        fill_f32<<<gb(nh), 256, 0, stream>>>(mb, -1e30f, nh);

        long waves = (long)(NNODES / 16) * (l.odim / 64);   // 16x64 tile per wave
        gemm_wmma_bf16<<<gb(waves * 32), 256, 0, stream>>>(xb, Wt, hbuf, NNODES, l.din, l.odim);

        attn_coef<<<gb(nh), 256, 0, stream>>>(hbuf, l.as, l.ad, asb, adb, l.heads);
        edge_max_k<<<gb(te), 256, 0, stream>>>(row, col, asb, adb, mb, l.heads);
        edge_expsum_k<<<gb(te), 256, 0, stream>>>(row, col, asb, adb, mb, sb, l.heads);
        edge_alpha_k<<<gb(te), 256, 0, stream>>>(row, col, asb, adb, mb, sb, alphab, l.heads);
        edge_agg_k<<<gb(te * HDIM), 256, 0, stream>>>(row, col, alphab, hbuf, agg, l.heads);

        float* dst = (li == 2) ? (float*)d_out : xf;
        finalize_ln_elu<<<NNODES, 256, 0, stream>>>(agg, l.b, l.g, l.be, dst, l.odim);
    }
}
